// TreeLSTM_67138928771801
// MI455X (gfx1250) — matrix operand order repaired and checked by reading, not compiled, not run on hardware
//
#include <hip/hip_runtime.h>
#include <hip/hip_bf16.h>

// ---------------------------------------------------------------------------
// TreeLSTM on MI455X (gfx1250): bf16 WMMA GEMMs, software-pipelined B-fragment
// prefetch, double-buffered LDS A staging, fused gate epilogue.
// ---------------------------------------------------------------------------

#define B_     32
#define S_     2048
#define E_     300
#define H_     256
#define STEPS_ 24
#define NIOU   (3 * H_)       // 768  : i|o|u columns
#define NX     (NIOU + H_)    // 1024 : x_iou | x_f
#define NG     (NIOU + 2*H_)  // 1280 : i|o|u|f_l|f_r gate columns
#define KU     (2 * H_)       // 512  : [h_l|h_r] contraction dim
#define KWPAD  320            // E=300 padded to multiple of 32
#define MROWS  (B_ * S_)      // 65536

typedef __bf16 bf16_t;
typedef __attribute__((ext_vector_type(16))) __bf16 v16bf;
typedef __attribute__((ext_vector_type(8)))  __bf16 v8bf;
typedef __attribute__((ext_vector_type(8)))  float  v8f;

static __device__ inline v8f wmma_bf16(v16bf a, v16bf b, v8f c) {
  // D = A(16x32) * B(32x16) + C(16x16), f32 accumulate
  return __builtin_amdgcn_wmma_f32_16x16x32_bf16(
      /*neg_a=*/false, a, /*neg_b=*/false, b,
      /*c_mod=*/(short)0, c, /*reuse_a=*/false, /*reuse_b=*/false);
}

// 16-bit A-matrix 16x32 fragment (contiguous k order, matches B):
//  lane<16 : halves 0..15 = K[0..15]    (row M = lane)
//  lane>=16: halves 0..15 = K[16..31]   (row M = lane-16)
static __device__ inline v16bf load_a_frag(const bf16_t* As /*16x32 row-major*/,
                                           int lane) {
  int row = lane & 15;
  int kg  = lane >> 4;
  const bf16_t* p = As + row * 32 + kg * 16;
  v8bf lo = *(const v8bf*)(p);        // K = kg*16 .. kg*16+7
  v8bf hi = *(const v8bf*)(p + 8);    // K = kg*16+8 .. kg*16+15
  return __builtin_shufflevector(lo, hi, 0, 1, 2, 3, 4, 5, 6, 7,
                                         8, 9, 10, 11, 12, 13, 14, 15);
}

// 16-bit B-matrix 32x16 fragment: lanes 0..15 hold K=0..15 of column N=lane,
// lanes 16..31 hold K=16..31 of column N=lane-16.  BT is [N][ldb] row-major
// (K-contiguous per N), so this is one 32-byte contiguous read per lane.
static __device__ inline v16bf load_b_frag(const bf16_t* __restrict__ BT_nbase,
                                           int lane, int kbase, int ldb) {
  int n  = lane & 15;
  int kg = lane >> 4;
  return *(const v16bf*)(BT_nbase + (size_t)n * ldb + kbase + kg * 16);
}

static __device__ inline float sigmoidf_(float x) {
  return 1.0f / (1.0f + __expf(-x));
}

// ---------------------------------------------------------------------------
// One-time weight repacks (f32 -> bf16, transposed to [N][K] K-contiguous)
// ---------------------------------------------------------------------------
__global__ void conv_u(const float* __restrict__ Ul,  const float* __restrict__ Ur,
                       const float* __restrict__ Fll, const float* __restrict__ Flr,
                       const float* __restrict__ Frl, const float* __restrict__ Frr,
                       bf16_t* __restrict__ UcatT) {
  int n = blockIdx.x;                       // 0..1279
  for (int k = threadIdx.x; k < KU; k += blockDim.x) {
    float v;
    if (k < H_) {                           // h_l weights
      if (n < NIOU)     v = Ul[(size_t)k * NIOU + n];
      else if (n < NX)  v = Fll[(size_t)k * H_ + (n - NIOU)];
      else              v = Frl[(size_t)k * H_ + (n - NX)];
    } else {                                // h_r weights
      int kr = k - H_;
      if (n < NIOU)     v = Ur[(size_t)kr * NIOU + n];
      else if (n < NX)  v = Flr[(size_t)kr * H_ + (n - NIOU)];
      else              v = Frr[(size_t)kr * H_ + (n - NX)];
    }
    UcatT[(size_t)n * KU + k] = (bf16_t)v;
  }
}

__global__ void conv_w(const float* __restrict__ Wiou,
                       const float* __restrict__ Wf,
                       bf16_t* __restrict__ WcatT) {
  int n = blockIdx.x;                       // 0..1023
  for (int k = threadIdx.x; k < KWPAD; k += blockDim.x) {
    float v = 0.0f;
    if (k < E_)
      v = (n < NIOU) ? Wiou[(size_t)k * NIOU + n]
                     : Wf[(size_t)k * H_ + (n - NIOU)];
    WcatT[(size_t)n * KWPAD + k] = (bf16_t)v;
  }
}

// ---------------------------------------------------------------------------
// Precompute:  Xcat[m, 0:768]  = X@W_iou + b_iou
//              Xcat[m,768:1024]= X@W_f   + b_f
// M-tile = 16 rows, 8 waves x 128 cols = N 1024. K padded 320, chunks of 32.
// Double-buffered A staging; B fragments prefetched one chunk ahead.
// ---------------------------------------------------------------------------
__global__ __launch_bounds__(256) void xproj_kernel(
    const float*  __restrict__ X,       // [MROWS, 300]
    const bf16_t* __restrict__ WcatT,   // [1024][320]
    const float*  __restrict__ b_iou,   // [768]
    const float*  __restrict__ b_f,     // [256]
    float*        __restrict__ Xcat) {  // [MROWS, 1024]
  __shared__ bf16_t As[2][16 * 32];
  const int rowBase = blockIdx.x * 16;
  const int tid  = threadIdx.x;
  const int wave = tid >> 5;            // 0..7
  const int lane = tid & 31;
  const int nbase = wave * 128;
  const int NCH = KWPAD / 32;           // 10 chunks

  auto stageA = [&](int kb, int slot) {
    int m  = tid >> 4;
    int kk = (tid & 15) * 2;
    int k0 = kb + kk;
    float f0 = (k0     < E_) ? X[(size_t)(rowBase + m) * E_ + k0]     : 0.f;
    float f1 = (k0 + 1 < E_) ? X[(size_t)(rowBase + m) * E_ + k0 + 1] : 0.f;
    As[slot][m * 32 + kk]     = (bf16_t)f0;
    As[slot][m * 32 + kk + 1] = (bf16_t)f1;
  };

  const v8f vzero = {0.f, 0.f, 0.f, 0.f, 0.f, 0.f, 0.f, 0.f};
  v8f acc[8];
#pragma unroll
  for (int i = 0; i < 8; ++i) acc[i] = vzero;

  stageA(0, 0);
  __syncthreads();

  v16bf bfrag[8];
#pragma unroll
  for (int ns = 0; ns < 8; ++ns)
    bfrag[ns] = load_b_frag(WcatT + (size_t)(nbase + ns * 16) * KWPAD,
                            lane, 0, KWPAD);

  int buf = 0;
  for (int ch = 0; ch < NCH; ++ch) {
    int kb = ch * 32;
    if (ch + 1 < NCH) stageA(kb + 32, buf ^ 1);   // overlap with WMMAs
    v16bf a = load_a_frag(As[buf], lane);
    v16bf bcur[8];
#pragma unroll
    for (int ns = 0; ns < 8; ++ns) bcur[ns] = bfrag[ns];
    if (ch + 1 < NCH) {
#pragma unroll
      for (int ns = 0; ns < 8; ++ns)              // prefetch next-chunk B
        bfrag[ns] = load_b_frag(WcatT + (size_t)(nbase + ns * 16) * KWPAD,
                                lane, kb + 32, KWPAD);
    }
#pragma unroll
    for (int ns = 0; ns < 8; ++ns)
      acc[ns] = wmma_bf16(a, bcur[ns], acc[ns]);
    __syncthreads();
    buf ^= 1;
  }

  // Epilogue: C/D layout — VGPR j holds row M=(lane>>4)*8+j, col N=lane&15.
  const int colLane = lane & 15;
  const int rsel    = lane >> 4;
#pragma unroll
  for (int ns = 0; ns < 8; ++ns) {
    int col  = nbase + ns * 16 + colLane;
    float bias = (col < NIOU) ? b_iou[col] : b_f[col - NIOU];
#pragma unroll
    for (int j = 0; j < 8; ++j) {
      int m = rsel * 8 + j;
      Xcat[(size_t)(rowBase + m) * NX + col] = acc[ns][j] + bias;
    }
  }
}

// ---------------------------------------------------------------------------
// Per-step fused kernel: gather h_l/h_r -> bf16 WMMA GEMM [16x512]x[512x1280]
// -> +Xcat -> gates in LDS -> sigmoid/tanh -> c/h update -> scatter.
// 10 waves (320 threads): full gate row (1280 cols) resident per workgroup.
// ---------------------------------------------------------------------------
__global__ __launch_bounds__(320) void step_kernel(
    const float* __restrict__ h_cur, const float* __restrict__ c_cur,
    float* __restrict__ h_next,      float* __restrict__ c_next,
    const float*  __restrict__ Xcat,   // [MROWS, 1024]
    const bf16_t* __restrict__ UcatT,  // [1280][512]
    const int* __restrict__ idd_all,   // [B, STEPS, S]
    const int* __restrict__ idl_all,
    const int* __restrict__ idr_all,
    int t) {
  __shared__ bf16_t As[2][16 * 32];    // double-buffered A stage (2 KB)
  __shared__ float  G[16 * NG];        // gate buffer (80 KB)
  __shared__ int    sIdd[16], sIdl[16], sIdr[16];

  const int rowBase = blockIdx.x * 16;          // flat (b,s) row
  const int b  = rowBase / S_;
  const int s0 = rowBase % S_;
  const int tid  = threadIdx.x;
  const int wave = tid >> 5;                    // 0..9
  const int lane = tid & 31;
  const int nbase = wave * 128;
  const int NCH = KU / 32;                      // 16 chunks

  if (tid < 16) {
    size_t base = ((size_t)b * STEPS_ + t) * S_ + s0 + tid;
    sIdd[tid] = idd_all[base];
    sIdl[tid] = idl_all[base];
    sIdr[tid] = idr_all[base];
  }
  __syncthreads();

  const float* hb = h_cur + (size_t)b * S_ * H_;

  auto stageA = [&](int kb, int slot) {
    if (tid < 256) {
      int m  = tid >> 4;
      int kk = (tid & 15) * 2;
      int k  = kb + kk;     // chunk lies entirely in h_l (k<256) or h_r half
      const float* src = (k < H_)
          ? hb + (size_t)sIdl[m] * H_ + k
          : hb + (size_t)sIdr[m] * H_ + (k - H_);
      float2 f = *(const float2*)src;
      As[slot][m * 32 + kk]     = (bf16_t)f.x;
      As[slot][m * 32 + kk + 1] = (bf16_t)f.y;
    }
  };

  const v8f vzero = {0.f, 0.f, 0.f, 0.f, 0.f, 0.f, 0.f, 0.f};
  v8f acc[8];
#pragma unroll
  for (int i = 0; i < 8; ++i) acc[i] = vzero;

  stageA(0, 0);
  __syncthreads();

  v16bf bfrag[8];
#pragma unroll
  for (int ns = 0; ns < 8; ++ns)
    bfrag[ns] = load_b_frag(UcatT + (size_t)(nbase + ns * 16) * KU,
                            lane, 0, KU);

  int buf = 0;
  for (int ch = 0; ch < NCH; ++ch) {
    int kb = ch * 32;
    if (ch + 1 < NCH) stageA(kb + 32, buf ^ 1);   // overlap with WMMAs
    v16bf a = load_a_frag(As[buf], lane);
    v16bf bcur[8];
#pragma unroll
    for (int ns = 0; ns < 8; ++ns) bcur[ns] = bfrag[ns];
    if (ch + 1 < NCH) {
#pragma unroll
      for (int ns = 0; ns < 8; ++ns)              // prefetch next-chunk B
        bfrag[ns] = load_b_frag(UcatT + (size_t)(nbase + ns * 16) * KU,
                                lane, kb + 32, KU);
    }
#pragma unroll
    for (int ns = 0; ns < 8; ++ns)
      acc[ns] = wmma_bf16(a, bcur[ns], acc[ns]);
    __syncthreads();
    buf ^= 1;
  }

  // Accum + Xcat -> LDS gate buffer. Gate col -> x col: iou/f_l map 1:1,
  // f_r (>=1024) maps to x_f block at col-256.
  const int colLane = lane & 15;
  const int rsel    = lane >> 4;
#pragma unroll
  for (int ns = 0; ns < 8; ++ns) {
    int col  = nbase + ns * 16 + colLane;
    int xcol = (col < NX) ? col : (col - H_);
#pragma unroll
    for (int j = 0; j < 8; ++j) {
      int m = rsel * 8 + j;
      float x = Xcat[(size_t)(rowBase + m) * NX + xcol];
      G[m * NG + col] = acc[ns][j] + x;
    }
  }
  __syncthreads();

  // Fused gate math + scatter (16 rows x 256 hidden units).
  const float* cb  = c_cur  + (size_t)b * S_ * H_;
  float*       hnb = h_next + (size_t)b * S_ * H_;
  float*       cnb = c_next + (size_t)b * S_ * H_;
  for (int e = tid; e < 16 * H_; e += blockDim.x) {
    int m = e >> 8;
    int j = e & (H_ - 1);
    float i_g = sigmoidf_(G[m * NG + j]);
    float o_g = sigmoidf_(G[m * NG + H_ + j]);
    float u_g = tanhf(G[m * NG + 2 * H_ + j]);
    float fl  = sigmoidf_(G[m * NG + NIOU + j]);
    float fr  = sigmoidf_(G[m * NG + NX + j]);
    float cl  = cb[(size_t)sIdl[m] * H_ + j];
    float cr  = cb[(size_t)sIdr[m] * H_ + j];
    float cn  = i_g * u_g + fl * cl + fr * cr;
    float hn  = o_g * tanhf(cn);
    int d = sIdd[m];
    cnb[(size_t)d * H_ + j] = cn;
    hnb[(size_t)d * H_ + j] = hn;
  }
}

// ---------------------------------------------------------------------------
__global__ void zero_slot0(float* __restrict__ h, float* __restrict__ c) {
  int b = blockIdx.x;
  int j = threadIdx.x;  // 256
  h[(size_t)b * S_ * H_ + j] = 0.0f;
  c[(size_t)b * S_ * H_ + j] = 0.0f;
}

__global__ void root_kernel(const int* __restrict__ tree_ids,
                            const float* __restrict__ h,
                            float* __restrict__ out_root) {
  __shared__ int red[256];
  int b = blockIdx.x;
  const int* ids = tree_ids + ((size_t)b * STEPS_ + (STEPS_ - 1)) * S_;
  int mx = 0;
  for (int s = threadIdx.x; s < S_; s += 256) mx = max(mx, ids[s]);
  red[threadIdx.x] = mx;
  __syncthreads();
  for (int off = 128; off > 0; off >>= 1) {
    if (threadIdx.x < off)
      red[threadIdx.x] = max(red[threadIdx.x], red[threadIdx.x + off]);
    __syncthreads();
  }
  int root = red[0];
  out_root[(size_t)b * H_ + threadIdx.x] =
      h[((size_t)b * S_ + root) * H_ + threadIdx.x];
}

// ---------------------------------------------------------------------------
extern "C" void kernel_launch(void* const* d_in, const int* in_sizes, int n_in,
                              void* d_out, int out_size, void* d_ws,
                              size_t ws_size, hipStream_t stream) {
  (void)in_sizes; (void)n_in; (void)out_size; (void)ws_size;

  const float* input_ids  = (const float*)d_in[0];
  const int*   tree_ids   = (const int*)d_in[1];
  const int*   tree_ids_r = (const int*)d_in[2];
  const int*   tree_ids_l = (const int*)d_in[3];
  const float* W_iou      = (const float*)d_in[4];
  const float* b_iou      = (const float*)d_in[5];
  const float* U_iou_l    = (const float*)d_in[6];
  const float* U_iou_r    = (const float*)d_in[7];
  const float* W_f        = (const float*)d_in[8];
  const float* b_f        = (const float*)d_in[9];
  const float* U_f_ll     = (const float*)d_in[10];
  const float* U_f_lr     = (const float*)d_in[11];
  const float* U_f_rl     = (const float*)d_in[12];
  const float* U_f_rr     = (const float*)d_in[13];

  const size_t HC = (size_t)B_ * S_ * H_ * sizeof(float);  // 64 MB
  char* ws = (char*)d_ws;
  float* h0 = (float*)ws;          ws += HC;
  float* h1 = (float*)ws;          ws += HC;
  float* c0 = (float*)ws;          ws += HC;
  float* c1 = (float*)ws;          ws += HC;
  float* Xcat = (float*)ws;        ws += (size_t)MROWS * NX * sizeof(float);
  bf16_t* UcatT = (bf16_t*)ws;     ws += (size_t)NG * KU * sizeof(bf16_t);
  bf16_t* WcatT = (bf16_t*)ws;     ws += (size_t)NX * KWPAD * sizeof(bf16_t);

  // h0/c0 = zeros (step 0 state)
  hipMemsetAsync(h0, 0, HC, stream);
  hipMemsetAsync(c0, 0, HC, stream);

  // One-time weight repacks.
  conv_u<<<NG, 256, 0, stream>>>(U_iou_l, U_iou_r, U_f_ll, U_f_lr,
                                 U_f_rl, U_f_rr, UcatT);
  conv_w<<<NX, 256, 0, stream>>>(W_iou, W_f, WcatT);

  // Step-invariant input projection.
  xproj_kernel<<<MROWS / 16, 256, 0, stream>>>(input_ids, WcatT, b_iou, b_f,
                                               Xcat);

  // 24 recurrence steps, double-buffered (copy-then-scatter semantics).
  float *hc = h0, *cc = c0, *hn = h1, *cn = c1;
  for (int t = 0; t < STEPS_; ++t) {
    hipMemcpyAsync(hn, hc, HC, hipMemcpyDeviceToDevice, stream);
    hipMemcpyAsync(cn, cc, HC, hipMemcpyDeviceToDevice, stream);
    step_kernel<<<MROWS / 16, 320, 0, stream>>>(
        hc, cc, hn, cn, Xcat, UcatT, tree_ids, tree_ids_l, tree_ids_r, t);
    zero_slot0<<<B_, 256, 0, stream>>>(hn, cn);
    float* th = hc; hc = hn; hn = th;
    float* tc = cc; cc = cn; cn = tc;
  }

  // Outputs: h | c | h_root  (flat, in return order).
  hipMemcpyAsync(d_out, hc, HC, hipMemcpyDeviceToDevice, stream);
  hipMemcpyAsync((char*)d_out + HC, cc, HC, hipMemcpyDeviceToDevice, stream);
  root_kernel<<<B_, 256, 0, stream>>>(tree_ids, hc,
                                      (float*)d_out + 2 * (size_t)B_ * S_ * H_);
}
